// BinaryTreeLSTM_91044716741116
// MI455X (gfx1250) — compile-verified, matrix operand order checked by reading, not run on hardware
//
#include <hip/hip_runtime.h>

// BinaryTreeLSTM pipeline for MI455X (gfx1250), wave32 + WMMA bf16 + TDM.
//
// Stages (all on `stream`):
//   0. pack: emb -> bf16; all weights -> bf16 WMMA-fragment layout
//   1. gx_{f,b}[t][4D][B] = (emb[x] @ W_ih^T + b), stored column-major per step so
//      the scan's accumulator init is one 16B load per lane (C-frag = fixed col,
//      consecutive rows)
//   2. LSTM scans: per-workgroup 16-batch-row tile, W_hh bf16 staged into LDS by the
//      Tensor Data Mover (tensor_load_to_lds + s_wait_tensorcnt), h ping-pong in LDS,
//      c in registers, 16 wmma/step/wave, 512 steps
//   3. leaves = [hf|hb] @ W_proj^T (WMMA GEMM) -> d_out + bf16 copy
//   4. gpre[t][5D][B] = leaves @ W_tree[:,D:]^T + b_tree (right child c2==0)
//   5. tree scan: gates = gpre + h1 @ W_tree[:,:D]^T; f2 gate skipped (c2==0);
//      W1 (160KB) TDM-staged in LDS, 16 wmma/step/wave, 511 steps

#define DEV __device__ __forceinline__

typedef __attribute__((ext_vector_type(16))) __bf16 v16bf;
typedef __attribute__((ext_vector_type(8)))  __bf16 v8bf;
typedef __attribute__((ext_vector_type(8)))  float  v8f;
typedef __attribute__((ext_vector_type(4)))  unsigned int v4u;
typedef __attribute__((ext_vector_type(4)))  int v4i;
typedef __attribute__((ext_vector_type(8)))  int v8i;

constexpr int BN = 256;          // batch
constexpr int LN = 512;          // sequence length
constexpr int DN = 128;          // hidden size
constexpr int MN = BN * LN;      // GEMM M dimension (time-major: m = t*BN + b)

DEV v8f wmma_bf16(v16bf a, v16bf b, v8f c) {
  // D = A(16x32 bf16) * B(32x16 bf16) + C(16x16 f32)
  return __builtin_amdgcn_wmma_f32_16x16x32_bf16(false, a, false, b, (short)0, c,
                                                 false, false);
}

// Two contiguous 16B chunks -> one 16-element bf16 fragment register set.
DEV v16bf frag2(const __bf16* p0, const __bf16* p1) {
  v8bf lo = *(const v8bf*)p0;
  v8bf hi = *(const v8bf*)p1;
  return __builtin_shufflevector(lo, hi, 0, 1, 2, 3, 4, 5, 6, 7,
                                 8, 9, 10, 11, 12, 13, 14, 15);
}

DEV float sigf(float x) { return 1.0f / (1.0f + __expf(-x)); }
DEV float tanha(float x) {
  float a = fabsf(x);
  float e = __expf(-2.0f * a);
  float t = (1.0f - e) / (1.0f + e);
  return copysignf(t, x);
}

// Tensor Data Mover: 1-D DMA of `bytes` (multiple of 8, <=512KB) from global
// memory to LDS offset `lds_off` of this workgroup's allocation.
// D# layout per CDNA5 ISA sec.8: group0 {count=1, lds_addr,
// global_addr, type=2}; group1 {data_size=3 (8B), tensor_dim0, tensor_dim1=1,
// tile_dim0, tile_dim1=1, tensor_dim0_stride}.
DEV void tdm_load_to_lds(unsigned lds_off, const void* gaddr, unsigned bytes) {
  const unsigned long long ga = (unsigned long long)gaddr;
  const unsigned elems = bytes >> 3;   // 8-byte elements; tile_dim0 is 16-bit
  v4u g0;
  g0[0] = 1u;                                       // count=1 (valid user D#)
  g0[1] = lds_off;                                  // lds_addr [63:32]
  g0[2] = (unsigned)(ga & 0xffffffffu);             // global_addr [95:64]
  g0[3] = (unsigned)(ga >> 32) | (2u << 30);        // addr hi | type=2 [127:126]
  v8i g1;
  g1[0] = 3 << 16;                                  // data_size=8B, no multicast
  g1[1] = (int)((elems & 0xffffu) << 16);           // tensor_dim0[15:0] @ bits63:48
  g1[2] = (int)(((elems >> 16) & 0xffffu) | (1u << 16)); // dim0 hi | tensor_dim1=1
  g1[3] = (int)((elems & 0xffffu) << 16);           // tile_dim0 @ bits 127:112
  g1[4] = 1;                                        // tile_dim1 = 1
  g1[5] = (int)elems;                               // tensor_dim0_stride
  g1[6] = 0;
  g1[7] = 0;
  v4i z4 = {0, 0, 0, 0};
#if defined(__clang_major__) && __clang_major__ >= 23
  v8i z8 = {0, 0, 0, 0, 0, 0, 0, 0};
  __builtin_amdgcn_tensor_load_to_lds(g0, g1, z4, z4, z8, 0);
#else
  __builtin_amdgcn_tensor_load_to_lds(g0, g1, z4, z4, 0);
#endif
  __builtin_amdgcn_s_wait_tensorcnt(0);
}

// ---------------------------------------------------------------- pack kernels
__global__ void k_cvt_bf16(const float* __restrict__ s, __bf16* __restrict__ d, int n) {
  for (int i = blockIdx.x * blockDim.x + threadIdx.x; i < n;
       i += gridDim.x * blockDim.x)
    d[i] = (__bf16)s[i];
}

// Pack W ([N, Kstride] row-major, used as B[k][n] = W[n][kOffset+k]) into
// WMMA B-fragment order: [(kt*NT+nt)][lane][j] with 32B contiguous per lane.
__global__ void k_pack_frag(const float* __restrict__ W, __bf16* __restrict__ dst,
                            int NT, int total, int Kstride, int kOffset) {
  for (int i = blockIdx.x * blockDim.x + threadIdx.x; i < total;
       i += gridDim.x * blockDim.x) {
    int j    = i & 15;
    int lane = (i >> 4) & 31;
    int tile = i >> 9;
    int kt = tile / NT, nt = tile % NT;
    int kk = ((j < 8) ? j : j + 8) + (lane >> 4) * 8;
    int k  = kt * 32 + kk;
    int n  = nt * 16 + (lane & 15);
    dst[i] = (__bf16)W[(size_t)n * Kstride + kOffset + k];
  }
}

// -------------------------------------------- stage 1: gx = emb[x] @ W_ih^T + b
// Output layout: gx[t][col(4D)][b(B)] so the scan reads one v8bf per C-fragment.
__global__ void __launch_bounds__(256)
k_embed_gemm(const long long* __restrict__ x, const __bf16* __restrict__ embb,
             const __bf16* __restrict__ wpk, const float* __restrict__ bias,
             __bf16* __restrict__ gx) {
  const int tid = threadIdx.x, wv = tid >> 5, lane = tid & 31;
  const int khalf = lane >> 4, l15 = lane & 15;
  const int m0 = blockIdx.x * 16;
  const int m = m0 + l15;            // A-matrix row for this lane
  const int t = m >> 8, b = m & 255; // m = t*BN + b; all 16 rows share t
  const int b0c = m0 & 255;
  const long long idx = x[(size_t)b * LN + t];
  const __bf16* arow = embb + (size_t)idx * DN;

  v16bf a[4];
#pragma unroll
  for (int kt = 0; kt < 4; ++kt) {
    const __bf16* base = arow + kt * 32 + khalf * 8;
    a[kt] = frag2(base, base + 16);
  }

#pragma unroll
  for (int i = 0; i < 4; ++i) {
    const int nt = i * 8 + wv;
    const float bv = bias[nt * 16 + l15]; // bias depends only on N -> fill C frag
    v8f acc;
#pragma unroll
    for (int r = 0; r < 8; ++r) acc[r] = bv;
#pragma unroll
    for (int kt = 0; kt < 4; ++kt) {
      const __bf16* bb = wpk + ((size_t)(kt * 32 + nt) * 32 + lane) * 16;
      acc = wmma_bf16(a[kt], frag2(bb, bb + 8), acc);
    }
    // column-major store: one 16B store per lane
    const int col = nt * 16 + l15;
    v8bf ov;
#pragma unroll
    for (int r = 0; r < 8; ++r) ov[r] = (__bf16)acc[r];
    *(v8bf*)(gx + ((size_t)t * 512 + col) * BN + b0c + 8 * khalf) = ov;
  }
}

// ------------------------------------------------- stage 2: sequential LSTM scan
// grid = 16 blocks (one per 16 batch rows), 256 threads (8 waves).
// Wave w owns features [16w,16w+16): gate n-tiles {w, 8+w, 16+w, 24+w} = i,f,g,o.
__global__ void __launch_bounds__(256)
k_lstm_scan(const __bf16* __restrict__ gx, const __bf16* __restrict__ whh_pk,
            __bf16* __restrict__ h_out, int reverse) {
  extern __shared__ __align__(16) char smem[];
  __bf16* w_lds = (__bf16*)smem;               // 4kt*32nt*512 = 65536 bf16 (128KB)
  __bf16* hbuf0 = (__bf16*)(smem + 131072);    // 16x128 bf16 ping
  __bf16* hbuf1 = hbuf0 + 16 * 128;            // 16x128 bf16 pong

  const int tid = threadIdx.x, wv = tid >> 5, lane = tid & 31;
  const int khalf = lane >> 4, l15 = lane & 15;
  const int b0 = blockIdx.x * 16;
  const int feat = wv * 16 + l15;

  if (wv == 0) // TDM: DMA W_hh fragments into LDS (w_lds is at LDS offset 0)
    tdm_load_to_lds(0u, whh_pk, 131072u);
  for (int i = tid; i < 2 * 16 * 128; i += 256) hbuf0[i] = (__bf16)0.0f; // h(-1)=0
  __syncthreads();

  v8f c;
#pragma unroll
  for (int r = 0; r < 8; ++r) c[r] = 0.0f;

  for (int t = 0; t < LN; ++t) {
    const int tm = reverse ? (LN - 1 - t) : t;
    const __bf16* hcur = (t & 1) ? hbuf1 : hbuf0;
    __bf16* hnxt = (t & 1) ? hbuf0 : hbuf1;

    // accumulators init = precomputed input gates (one 16B load per gate tile)
    v8f g[4];
#pragma unroll
    for (int gi = 0; gi < 4; ++gi) {
      const int col = gi * 128 + feat;
      v8bf gv = *(const v8bf*)(gx + ((size_t)tm * 512 + col) * BN + b0 + 8 * khalf);
#pragma unroll
      for (int r = 0; r < 8; ++r) g[gi][r] = (float)gv[r];
    }

    // A fragments of h(t-1) from LDS, reused across the wave's 4 gate tiles
    v16bf a[4];
#pragma unroll
    for (int kt = 0; kt < 4; ++kt) {
      const __bf16* base = hcur + l15 * 128 + kt * 32 + khalf * 8;
      a[kt] = frag2(base, base + 16);
    }

    // g += h @ W_hh^T  : 16 wmma per step per wave
#pragma unroll
    for (int gi = 0; gi < 4; ++gi) {
      const int nt = gi * 8 + wv;
#pragma unroll
      for (int kt = 0; kt < 4; ++kt) {
        const __bf16* bb = w_lds + ((size_t)(kt * 32 + nt) * 32 + lane) * 16;
        g[gi] = wmma_bf16(a[kt], frag2(bb, bb + 8), g[gi]);
      }
    }

    // cell update (PyTorch gate order i,f,g,o)
    v8f h;
#pragma unroll
    for (int r = 0; r < 8; ++r) {
      float it = sigf(g[0][r]);
      float ft = sigf(g[1][r]);
      float gt = tanha(g[2][r]);
      float ot = sigf(g[3][r]);
      c[r] = ft * c[r] + it * gt;
      h[r] = ot * tanha(c[r]);
    }

    // h -> LDS (next step A) + global bf16 (projection GEMM A); stores don't
    // block the recurrence (STOREcnt, no RAW in this wave)
#pragma unroll
    for (int r = 0; r < 8; ++r) {
      const int row = r + 8 * khalf;
      __bf16 hbv = (__bf16)h[r];
      hnxt[row * 128 + feat] = hbv;
      h_out[((size_t)tm * BN + b0 + row) * 128 + feat] = hbv;
    }

    if (t + 1 < LN) { // hide next step's gx fetch -> global_prefetch_b8
      const int tn = reverse ? (LN - 2 - t) : (t + 1);
      __builtin_prefetch(&gx[((size_t)tn * 512 + feat) * BN + b0], 0, 1);
    }
    __syncthreads();
  }
}

// ------------------------------------- stage 3: leaves = [hf|hb] @ W_proj^T
__global__ void __launch_bounds__(256)
k_proj(const __bf16* __restrict__ hf, const __bf16* __restrict__ hb,
       const __bf16* __restrict__ wpk, float* __restrict__ out,
       __bf16* __restrict__ leaves) {
  const int tid = threadIdx.x, wv = tid >> 5, lane = tid & 31;
  const int khalf = lane >> 4, l15 = lane & 15;
  const int m0 = blockIdx.x * 16;
  const int m = m0 + l15;

  v16bf a[8]; // K = 256: kt 0..3 from hf, 4..7 from hb
#pragma unroll
  for (int kt = 0; kt < 8; ++kt) {
    const __bf16* src = (kt < 4) ? hf : hb;
    const __bf16* base = src + (size_t)m * DN + (kt & 3) * 32 + khalf * 8;
    a[kt] = frag2(base, base + 16);
  }

  const int nt = wv; // N = 128 -> 8 tiles, one per wave
  v8f acc;
#pragma unroll
  for (int r = 0; r < 8; ++r) acc[r] = 0.0f;
#pragma unroll
  for (int kt = 0; kt < 8; ++kt) {
    const __bf16* bb = wpk + ((size_t)(kt * 8 + nt) * 32 + lane) * 16;
    acc = wmma_bf16(a[kt], frag2(bb, bb + 8), acc);
  }

#pragma unroll
  for (int r = 0; r < 8; ++r) {
    const int row = m0 + r + 8 * khalf;
    const int tt = row >> 8, bb = row & 255;
    const int col = nt * 16 + l15;
    out[((size_t)bb * (2 * LN - 1) + tt) * DN + col] = acc[r]; // leaves -> output
    leaves[(size_t)row * DN + col] = (__bf16)acc[r];
  }
}

// ------------------- stage 4: gpre = leaves @ W_tree[:,D:2D]^T + b_tree (c2==0)
// Output layout: gpre[t][col(5D)][b(B)] (column-major per step, as with gx).
__global__ void __launch_bounds__(256)
k_tree_pre(const __bf16* __restrict__ leaves, const __bf16* __restrict__ wpk,
           const float* __restrict__ btree, __bf16* __restrict__ gpre) {
  const int tid = threadIdx.x, wv = tid >> 5, lane = tid & 31;
  const int khalf = lane >> 4, l15 = lane & 15;
  const int m0 = blockIdx.x * 16;
  const int m = m0 + l15;
  const int t = m0 >> 8, b0c = m0 & 255;
  const __bf16* arow = leaves + (size_t)m * DN;

  v16bf a[4];
#pragma unroll
  for (int kt = 0; kt < 4; ++kt) {
    const __bf16* base = arow + kt * 32 + khalf * 8;
    a[kt] = frag2(base, base + 16);
  }

#pragma unroll
  for (int i = 0; i < 5; ++i) { // N = 640 -> 40 tiles, 5 per wave
    const int nt = i * 8 + wv;
    const float bv = btree[nt * 16 + l15];
    v8f acc;
#pragma unroll
    for (int r = 0; r < 8; ++r) acc[r] = bv;
#pragma unroll
    for (int kt = 0; kt < 4; ++kt) {
      const __bf16* bb = wpk + ((size_t)(kt * 40 + nt) * 32 + lane) * 16;
      acc = wmma_bf16(a[kt], frag2(bb, bb + 8), acc);
    }
    const int col = nt * 16 + l15;
    v8bf ov;
#pragma unroll
    for (int r = 0; r < 8; ++r) ov[r] = (__bf16)acc[r];
    *(v8bf*)(gpre + ((size_t)t * 640 + col) * BN + b0c + 8 * khalf) = ov;
  }
}

// ------------------------------------------------- stage 5: tree-LSTM scan
// gates split i,f1,f2,o,g; f2 skipped (right child is a leaf: c2 == 0).
__global__ void __launch_bounds__(256)
k_tree_scan(const __bf16* __restrict__ gpre, const __bf16* __restrict__ wt1_pk,
            const __bf16* __restrict__ leaves, float* __restrict__ out) {
  extern __shared__ __align__(16) char smem[];
  __bf16* w_lds = (__bf16*)smem;               // 4kt*40nt*512 = 81920 bf16 (160KB)
  __bf16* hbuf0 = (__bf16*)(smem + 163840);
  __bf16* hbuf1 = hbuf0 + 16 * 128;

  const int tid = threadIdx.x, wv = tid >> 5, lane = tid & 31;
  const int khalf = lane >> 4, l15 = lane & 15;
  const int b0 = blockIdx.x * 16;
  const int feat = wv * 16 + l15;

  if (wv == 0) // TDM: DMA W_tree[:, :D] fragments into LDS
    tdm_load_to_lds(0u, wt1_pk, 163840u);
  // h1(0) = leaves[t=0] for this block's batch rows
  for (int i = tid; i < 16 * 128; i += 256)
    hbuf0[i] = leaves[((size_t)(b0 + (i >> 7))) * DN + (i & 127)];
  __syncthreads();

  const int ntg[4] = {wv, 8 + wv, 24 + wv, 32 + wv}; // i, f1, o, g
  v8f c;
#pragma unroll
  for (int r = 0; r < 8; ++r) c[r] = 0.0f;

  for (int t = 1; t < LN; ++t) {
    const __bf16* hcur = (t & 1) ? hbuf0 : hbuf1;
    __bf16* hnxt = (t & 1) ? hbuf1 : hbuf0;

    v8f g[4];
#pragma unroll
    for (int gi = 0; gi < 4; ++gi) {
      const int col = ntg[gi] * 16 + l15;
      v8bf gv = *(const v8bf*)(gpre + ((size_t)t * 640 + col) * BN + b0 + 8 * khalf);
#pragma unroll
      for (int r = 0; r < 8; ++r) g[gi][r] = (float)gv[r];
    }

    v16bf a[4];
#pragma unroll
    for (int kt = 0; kt < 4; ++kt) {
      const __bf16* base = hcur + l15 * 128 + kt * 32 + khalf * 8;
      a[kt] = frag2(base, base + 16);
    }

#pragma unroll
    for (int gi = 0; gi < 4; ++gi) {
#pragma unroll
      for (int kt = 0; kt < 4; ++kt) {
        const __bf16* bb = w_lds + ((size_t)(kt * 40 + ntg[gi]) * 32 + lane) * 16;
        g[gi] = wmma_bf16(a[kt], frag2(bb, bb + 8), g[gi]);
      }
    }

    v8f h;
#pragma unroll
    for (int r = 0; r < 8; ++r) {
      float it = sigf(g[0][r]);
      float f1 = sigf(g[1][r]);
      float ot = sigf(g[2][r]);
      float gt = tanha(g[3][r]);
      c[r] = f1 * c[r] + it * gt; // + sig(f2)*c2 omitted: c2 == 0
      h[r] = ot * tanha(c[r]);
    }

#pragma unroll
    for (int r = 0; r < 8; ++r) {
      const int row = r + 8 * khalf;
      const int bb = b0 + row;
      out[((size_t)bb * (2 * LN - 1) + (LN - 1 + t)) * DN + feat] = h[r];
      hnxt[row * 128 + feat] = (__bf16)h[r];
    }

    if (t + 1 < LN)
      __builtin_prefetch(&gpre[((size_t)(t + 1) * 640 + feat) * BN + b0], 0, 1);
    __syncthreads();
  }
}

// ---------------------------------------------------------------- orchestration
extern "C" void kernel_launch(void* const* d_in, const int* in_sizes, int n_in,
                              void* d_out, int out_size, void* d_ws, size_t ws_size,
                              hipStream_t stream) {
  (void)in_sizes; (void)n_in; (void)out_size; (void)ws_size;
  const long long* x  = (const long long*)d_in[0]; // reference dtype is int64
  const float* emb    = (const float*)d_in[1];
  const float* w_ih_f = (const float*)d_in[2];
  const float* w_hh_f = (const float*)d_in[3];
  const float* b_f    = (const float*)d_in[4];
  const float* w_ih_b = (const float*)d_in[5];
  const float* w_hh_b = (const float*)d_in[6];
  const float* b_b    = (const float*)d_in[7];
  const float* w_proj = (const float*)d_in[8];
  const float* w_tree = (const float*)d_in[9];
  const float* b_tree = (const float*)d_in[10];
  float* out = (float*)d_out;

  char* ws = (char*)d_ws;
  size_t off = 0;
  auto alloc = [&](size_t bytes) -> void* {
    void* p = ws + off;
    off += (bytes + 255) & ~(size_t)255;
    return p;
  };
  __bf16* emb_bf    = (__bf16*)alloc((size_t)32000 * DN * 2);
  __bf16* wihf_pk   = (__bf16*)alloc((size_t)65536 * 2);
  __bf16* wihb_pk   = (__bf16*)alloc((size_t)65536 * 2);
  __bf16* whhf_pk   = (__bf16*)alloc((size_t)65536 * 2);
  __bf16* whhb_pk   = (__bf16*)alloc((size_t)65536 * 2);
  __bf16* wproj_pk  = (__bf16*)alloc((size_t)32768 * 2);
  __bf16* wtree1_pk = (__bf16*)alloc((size_t)81920 * 2);
  __bf16* wtree2_pk = (__bf16*)alloc((size_t)81920 * 2);
  __bf16* gx_f      = (__bf16*)alloc((size_t)MN * 512 * 2);
  __bf16* gx_b      = (__bf16*)alloc((size_t)MN * 512 * 2);
  __bf16* gpre      = (__bf16*)alloc((size_t)MN * 640 * 2);
  __bf16* hf_bf     = (__bf16*)alloc((size_t)MN * DN * 2);
  __bf16* hb_bf     = (__bf16*)alloc((size_t)MN * DN * 2);
  __bf16* leaves_bf = (__bf16*)alloc((size_t)MN * DN * 2);

  // stage 0: pack
  k_cvt_bf16<<<4096, 256, 0, stream>>>(emb, emb_bf, 32000 * DN);
  k_pack_frag<<<256, 256, 0, stream>>>(w_ih_f, wihf_pk, 32, 65536, 128, 0);
  k_pack_frag<<<256, 256, 0, stream>>>(w_ih_b, wihb_pk, 32, 65536, 128, 0);
  k_pack_frag<<<256, 256, 0, stream>>>(w_hh_f, whhf_pk, 32, 65536, 128, 0);
  k_pack_frag<<<256, 256, 0, stream>>>(w_hh_b, whhb_pk, 32, 65536, 128, 0);
  k_pack_frag<<<128, 256, 0, stream>>>(w_proj, wproj_pk, 8, 32768, 256, 0);
  k_pack_frag<<<320, 256, 0, stream>>>(w_tree, wtree1_pk, 40, 81920, 256, 0);
  k_pack_frag<<<320, 256, 0, stream>>>(w_tree, wtree2_pk, 40, 81920, 256, 128);

  // stage 1: hoisted input GEMMs
  k_embed_gemm<<<MN / 16, 256, 0, stream>>>(x, emb_bf, wihf_pk, b_f, gx_f);
  k_embed_gemm<<<MN / 16, 256, 0, stream>>>(x, emb_bf, wihb_pk, b_b, gx_b);

  // stage 2: recurrent scans (batch-parallel, weights TDM-staged in LDS)
  k_lstm_scan<<<16, 256, 131072 + 2 * 4096, stream>>>(gx_f, whhf_pk, hf_bf, 0);
  k_lstm_scan<<<16, 256, 131072 + 2 * 4096, stream>>>(gx_b, whhb_pk, hb_bf, 1);

  // stage 3: projection -> leaves
  k_proj<<<MN / 16, 256, 0, stream>>>(hf_bf, hb_bf, wproj_pk, out, leaves_bf);

  // stage 4: hoisted leaf half of the tree gates
  k_tree_pre<<<MN / 16, 256, 0, stream>>>(leaves_bf, wtree2_pk, b_tree, gpre);

  // stage 5: tree scan -> internal nodes
  k_tree_scan<<<16, 256, 163840 + 2 * 4096, stream>>>(gpre, wtree1_pk, leaves_bf, out);
}
